// MultiModalGAT_76218489635507
// MI455X (gfx1250) — matrix operand order repaired and checked by reading, not compiled, not run on hardware
//
#include <hip/hip_runtime.h>
#include <hip/hip_bf16.h>

// ---------------------------------------------------------------------------
// MultiModalGAT for MI455X (gfx1250, wave32, WMMA)
//
// Roofline: mel GEMM reads 512MB f32 @ 23.3TB/s ~= 22us; 32.8 GFLOP at
// bf16 WMMA rates is far cheaper -> memory bound. Stream f32 once, convert to
// bf16 in LDS, v_wmma_f32_16x16x32_bf16 with f32 accumulate. K-step = 64:
// two WMMAs per barrier pair, float4 staging loads, ds_store_b64 to LDS.
// GAT scatter phase: h tables (20.5MB) live in 192MB L2 -> atomics are cheap.
// ---------------------------------------------------------------------------

typedef __attribute__((ext_vector_type(16))) __bf16 v16bf;
typedef __attribute__((ext_vector_type(8)))  __bf16 v8bf;
typedef __attribute__((ext_vector_type(4)))  __bf16 v4bf;
typedef __attribute__((ext_vector_type(8)))  float  v8f;

static __host__ __device__ inline long cdiv(long a, long b) { return (a + b - 1) / b; }

// -------------------- weight prep: f32 [K,N] -> bf16 [N,K] -----------------
__global__ void transpose_bf16(const float* __restrict__ W, __bf16* __restrict__ WT,
                               int K, int N) {
  long t = (long)blockIdx.x * blockDim.x + threadIdx.x;
  if (t >= (long)K * N) return;
  int k = (int)(t / N), n = (int)(t % N);
  WT[(long)n * K + k] = (__bf16)W[t];
}

// -------------------- bf16 WMMA GEMM ---------------------------------------
// C[M,N] = act( A[M,K](f32) @ WT[N,K](bf16)^T + bias )
// A source switches to A2 at column K1 (concat([text, mel_feat]) GEMM);
// both K1 and K-K1 are multiples of 64 for every call site.
// One wave = one 16x16 C tile. 8 waves/block share one 16-row A tile via LDS.

#define LDSW 72  // 64 + 8 halves pad -> 144B row stride, conflict-free b128

__device__ __forceinline__ void k_step64(
    const float* __restrict__ Ap, long ld, int kc, int kwt, bool pf,
    const __bf16* __restrict__ wrow, __bf16 (*As)[LDSW],
    int tid, int tileM, int arow, int kb, int kb16, v8f& acc) {
  __syncthreads();
  {
    // 256 threads x float4 = 1024 f32 = full 16x64 tile, coalesced b128
    int row  = tid >> 4;
    int cols = (tid & 15) * 4;
    const float* src = Ap + (long)(tileM + row) * ld + (kc + cols);
    float4 f = *(const float4*)src;
    v4bf h;
    h[0] = (__bf16)f.x; h[1] = (__bf16)f.y; h[2] = (__bf16)f.z; h[3] = (__bf16)f.w;
    *(v4bf*)&As[row][cols] = h;                       // ds_store_b64
    if (pf) __builtin_prefetch(src + 64, 0, 0);       // global_prefetch_b8
  }
  __syncthreads();
#pragma unroll
  for (int sub = 0; sub < 2; ++sub) {
    // A fragment: lane = row; K = kb+{0..15} contiguous
    v8bf alo = *(const v8bf*)&As[arow][sub * 32 + kb];        // ds_load_b128
    v8bf ahi = *(const v8bf*)&As[arow][sub * 32 + 8 + kb];
    v16bf afrag = __builtin_shufflevector(alo, ahi,
        0,1,2,3,4,5,6,7,8,9,10,11,12,13,14,15);
    // B fragment: 16 contiguous halves of WT row = one column of W
    v16bf bfrag = *(const v16bf*)(wrow + kwt + sub * 32 + kb16);
    acc = __builtin_amdgcn_wmma_f32_16x16x32_bf16(
        false, afrag, false, bfrag, (short)0, acc, false, false);
  }
}

__global__ __launch_bounds__(256) void gemm_bf16_wmma(
    const float* __restrict__ A, long lda,
    const float* __restrict__ A2, long lda2, int K1,
    const __bf16* __restrict__ WT,
    float* __restrict__ C, const float* __restrict__ bias,
    int N, int K, int act) {
  __shared__ __attribute__((aligned(32))) __bf16 As[16][LDSW];

  const int tid  = threadIdx.x;
  const int wave = tid >> 5;
  const int lane = tid & 31;
  const int tileM = blockIdx.x * 16;
  const int tileN = (blockIdx.y * 8 + wave) * 16;

  const int arow = lane & 15;
  const int kb   = (lane < 16) ? 0 : 16;   // A half-K offset (contiguous, matches B)
  const int kb16 = (lane < 16) ? 0 : 16;   // B half-K offset (contiguous, matches A)
  const __bf16* wrow = WT + (long)(tileN + (lane & 15)) * K;

  v8f acc = {};
  for (int k0 = 0; k0 < K1; k0 += 64)      // source A  (pointer select hoisted)
    k_step64(A, lda, k0, k0, k0 + 64 < K1, wrow, As, tid, tileM, arow, kb, kb16, acc);
  for (int k0 = K1; k0 < K; k0 += 64)      // source A2 (concat tail)
    k_step64(A2, lda2, k0 - K1, k0, k0 + 64 < K, wrow, As, tid, tileM, arow, kb, kb16, acc);

  const int ncol = lane & 15;
  float b = bias ? bias[tileN + ncol] : 0.0f;
#pragma unroll
  for (int j = 0; j < 8; ++j) {
    int mrow = j + ((lane >> 4) << 3);   // C/D layout: lanes 16-31 hold M=8..15
    float v = acc[j] + b;
    if (act == 1)      v = v > 0.0f ? v : 0.0f;              // relu
    else if (act == 2) v = v > 0.0f ? v : (expf(v) - 1.0f);  // elu
    C[(long)(tileM + mrow) * N + tileN + ncol] = v;
  }
}

// -------------------- GAT helpers ------------------------------------------
__device__ inline unsigned fmap(float f) {
  unsigned u = __float_as_uint(f);
  return (u >> 31) ? ~u : (u | 0x80000000u);
}
__device__ inline float funmap(unsigned u) {
  return (u >> 31) ? __uint_as_float(u & 0x7FFFFFFFu) : __uint_as_float(~u);
}
__device__ inline void edge_sd(const int* __restrict__ ei, long i, int E, int* s, int* d) {
  if (i < E) { *s = ei[i]; *d = ei[E + i]; }
  else       { *s = *d = (int)(i - E); }      // PyG self loops appended
}

// per-(node,head) attention logits + softmax-state init
__global__ void att_scores(const float* __restrict__ h, const float* __restrict__ ws,
                           const float* __restrict__ wd, float* __restrict__ a_s,
                           float* __restrict__ a_d, unsigned* __restrict__ mu,
                           float* __restrict__ ssum, int n, int H, int C) {
  long t = (long)blockIdx.x * blockDim.x + threadIdx.x;
  if (t >= (long)n * H) return;
  int node = (int)(t / H), hh = (int)(t % H);
  const float* hp = h + ((long)node * H + hh) * C;
  const float* pws = ws + (long)hh * C;
  const float* pwd = wd + (long)hh * C;
  float ss = 0.0f, sd = 0.0f;
  for (int c = 0; c < C; ++c) { float v = hp[c]; ss += v * pws[c]; sd += v * pwd[c]; }
  a_s[t] = ss; a_d[t] = sd;
  mu[t] = 0u;       // identity for mapped-float max
  ssum[t] = 0.0f;
}

__global__ void edge_max(const int* __restrict__ ei, int E, long total,
                         const float* __restrict__ a_s, const float* __restrict__ a_d,
                         unsigned* __restrict__ mu, int H) {
  long t = (long)blockIdx.x * blockDim.x + threadIdx.x;
  if (t >= total) return;
  long i = t / H; int hh = (int)(t % H);
  int s, d; edge_sd(ei, i, E, &s, &d);
  float v = a_s[(long)s * H + hh] + a_d[(long)d * H + hh];
  v = v > 0.0f ? v : 0.2f * v;                 // leaky_relu(0.2)
  atomicMax(&mu[(long)d * H + hh], fmap(v));
}

__global__ void edge_sum(const int* __restrict__ ei, int E, long total,
                         const float* __restrict__ a_s, const float* __restrict__ a_d,
                         const unsigned* __restrict__ mu, float* __restrict__ ssum, int H) {
  long t = (long)blockIdx.x * blockDim.x + threadIdx.x;
  if (t >= total) return;
  long i = t / H; int hh = (int)(t % H);
  int s, d; edge_sd(ei, i, E, &s, &d);
  float v = a_s[(long)s * H + hh] + a_d[(long)d * H + hh];
  v = v > 0.0f ? v : 0.2f * v;
  float m = funmap(mu[(long)d * H + hh]);
  atomicAdd(&ssum[(long)d * H + hh], expf(v - m));
}

__global__ void edge_acc(const int* __restrict__ ei, int E, long total,
                         const float* __restrict__ a_s, const float* __restrict__ a_d,
                         const unsigned* __restrict__ mu, const float* __restrict__ ssum,
                         const float* __restrict__ hsrc, float* __restrict__ outp,
                         int H, int C) {
  long t = (long)blockIdx.x * blockDim.x + threadIdx.x;
  if (t >= total) return;
  long i = t / H; int hh = (int)(t % H);
  int s, d; edge_sd(ei, i, E, &s, &d);
  float v = a_s[(long)s * H + hh] + a_d[(long)d * H + hh];
  v = v > 0.0f ? v : 0.2f * v;
  float m = funmap(mu[(long)d * H + hh]);
  float alpha = expf(v - m) / (ssum[(long)d * H + hh] + 1e-16f);
  const float* hp = hsrc + ((long)s * H + hh) * C;
  float*       op = outp + ((long)d * H + hh) * C;
  for (int c = 0; c < C; c += 4) {
    float4 hv = *(const float4*)(hp + c);
    atomicAdd(&op[c + 0], alpha * hv.x);
    atomicAdd(&op[c + 1], alpha * hv.y);
    atomicAdd(&op[c + 2], alpha * hv.z);
    atomicAdd(&op[c + 3], alpha * hv.w);
  }
}

__global__ void zero_f32(float* __restrict__ p, long total) {
  long t = (long)blockIdx.x * blockDim.x + threadIdx.x;
  if (t < total) p[t] = 0.0f;
}

__global__ void bias_elu(float* __restrict__ p, const float* __restrict__ b,
                         long total, int ncols) {
  long t = (long)blockIdx.x * blockDim.x + threadIdx.x;
  if (t >= total) return;
  float v = p[t] + b[t % ncols];
  p[t] = v > 0.0f ? v : (expf(v) - 1.0f);
}

// h2[n,4] = x1[n,256] @ g2_W[256,4]; also init out = g2_b (replay-safe)
__global__ void layer2_h(const float* __restrict__ x1, const float* __restrict__ W,
                         const float* __restrict__ b, float* __restrict__ h2,
                         float* __restrict__ out, int n) {
  int node = blockIdx.x * blockDim.x + threadIdx.x;
  if (node >= n) return;
  const float* xp = x1 + (long)node * 256;
  float a0 = 0, a1 = 0, a2 = 0, a3 = 0;
  for (int k = 0; k < 256; ++k) {
    float xv = xp[k];
    const float4 wr = *(const float4*)(W + (long)k * 4);
    a0 += xv * wr.x; a1 += xv * wr.y; a2 += xv * wr.z; a3 += xv * wr.w;
  }
  h2[(long)node * 4 + 0] = a0; h2[(long)node * 4 + 1] = a1;
  h2[(long)node * 4 + 2] = a2; h2[(long)node * 4 + 3] = a3;
  out[(long)node * 4 + 0] = b[0]; out[(long)node * 4 + 1] = b[1];
  out[(long)node * 4 + 2] = b[2]; out[(long)node * 4 + 3] = b[3];
}

// ---------------------------------------------------------------------------
extern "C" void kernel_launch(void* const* d_in, const int* in_sizes, int n_in,
                              void* d_out, int out_size, void* d_ws, size_t ws_size,
                              hipStream_t stream) {
  const float* text     = (const float*)d_in[0];
  const float* mel      = (const float*)d_in[1];
  const int*   ei       = (const int*)d_in[2];
  const float* mel_W    = (const float*)d_in[3];
  const float* mel_b    = (const float*)d_in[4];
  const float* concat_W = (const float*)d_in[5];
  const float* concat_b = (const float*)d_in[6];
  const float* g1_W     = (const float*)d_in[7];
  const float* g1_as    = (const float*)d_in[8];
  const float* g1_ad    = (const float*)d_in[9];
  const float* g1_b     = (const float*)d_in[10];
  const float* g2_W     = (const float*)d_in[11];
  const float* g2_as    = (const float*)d_in[12];
  const float* g2_ad    = (const float*)d_in[13];
  const float* g2_b     = (const float*)d_in[14];
  float* out = (float*)d_out;

  const int Nn = 20000, E = 640000, TXT = 768, MELK = 6400, HID = 128, H = 2;

  char* ws = (char*)d_ws;
  size_t off = 0;
  auto take = [&](size_t bytes) -> void* {
    void* p = ws + off;
    off = (off + bytes + 255) & ~(size_t)255;
    return p;
  };
  __bf16* wt1   = (__bf16*)take((size_t)HID * MELK * sizeof(__bf16));        // [128,6400]
  __bf16* wt2   = (__bf16*)take((size_t)HID * (TXT + HID) * sizeof(__bf16)); // [128,896]
  __bf16* wt3   = (__bf16*)take((size_t)(2 * HID) * HID * sizeof(__bf16));   // [256,128]
  float*  melF  = (float*)take((size_t)Nn * HID * 4);
  float*  x     = (float*)take((size_t)Nn * HID * 4);
  float*  h1    = (float*)take((size_t)Nn * 2 * HID * 4);
  float*  a1s   = (float*)take((size_t)Nn * H * 4);
  float*  a1d   = (float*)take((size_t)Nn * H * 4);
  unsigned* m1u = (unsigned*)take((size_t)Nn * H * 4);
  float*  s1    = (float*)take((size_t)Nn * H * 4);
  float*  x1    = (float*)take((size_t)Nn * 2 * HID * 4);
  float*  h2    = (float*)take((size_t)Nn * 4 * 4);
  float*  a2s   = (float*)take((size_t)Nn * 4);
  float*  a2d   = (float*)take((size_t)Nn * 4);
  unsigned* m2u = (unsigned*)take((size_t)Nn * 4);
  float*  s2    = (float*)take((size_t)Nn * 4);
  (void)ws_size; (void)n_in; (void)in_sizes; (void)out_size;

  const int B = 256;

  // weight prep -> bf16 transposed
  transpose_bf16<<<cdiv((long)MELK * HID, B), B, 0, stream>>>(mel_W, wt1, MELK, HID);
  transpose_bf16<<<cdiv((long)(TXT + HID) * HID, B), B, 0, stream>>>(concat_W, wt2, TXT + HID, HID);
  transpose_bf16<<<cdiv((long)HID * 2 * HID, B), B, 0, stream>>>(g1_W, wt3, HID, 2 * HID);

  // mel_feat = relu(mel @ mel_W + mel_b)        [20000,128]
  gemm_bf16_wmma<<<dim3(Nn / 16, 1), B, 0, stream>>>(
      mel, MELK, mel, MELK, MELK, wt1, melF, mel_b, HID, MELK, /*relu*/1);

  // x = elu(concat(text, mel_feat) @ concat_W + concat_b)   [20000,128]
  gemm_bf16_wmma<<<dim3(Nn / 16, 1), B, 0, stream>>>(
      text, TXT, melF, HID, TXT, wt2, x, concat_b, HID, TXT + HID, /*elu*/2);

  // h1 = x @ g1_W   [20000,256] (bias added after GAT aggregation)
  gemm_bf16_wmma<<<dim3(Nn / 16, 2), B, 0, stream>>>(
      x, HID, x, HID, HID, wt3, h1, nullptr, 2 * HID, HID, /*none*/0);

  // ---- GAT layer 1 ----
  att_scores<<<cdiv((long)Nn * H, B), B, 0, stream>>>(h1, g1_as, g1_ad, a1s, a1d, m1u, s1, Nn, H, HID);
  zero_f32<<<cdiv((long)Nn * 2 * HID, B), B, 0, stream>>>(x1, (long)Nn * 2 * HID);
  long tot1 = (long)(E + Nn) * H;
  edge_max<<<cdiv(tot1, B), B, 0, stream>>>(ei, E, tot1, a1s, a1d, m1u, H);
  edge_sum<<<cdiv(tot1, B), B, 0, stream>>>(ei, E, tot1, a1s, a1d, m1u, s1, H);
  edge_acc<<<cdiv(tot1, B), B, 0, stream>>>(ei, E, tot1, a1s, a1d, m1u, s1, h1, x1, H, HID);
  bias_elu<<<cdiv((long)Nn * 2 * HID, B), B, 0, stream>>>(x1, g1_b, (long)Nn * 2 * HID, 2 * HID);

  // ---- GAT layer 2 (heads=1, C=4) ----
  layer2_h<<<cdiv((long)Nn, B), B, 0, stream>>>(x1, g2_W, g2_b, h2, out, Nn);
  att_scores<<<cdiv((long)Nn, B), B, 0, stream>>>(h2, g2_as, g2_ad, a2s, a2d, m2u, s2, Nn, 1, 4);
  long tot2 = (long)(E + Nn);
  edge_max<<<cdiv(tot2, B), B, 0, stream>>>(ei, E, tot2, a2s, a2d, m2u, 1);
  edge_sum<<<cdiv(tot2, B), B, 0, stream>>>(ei, E, tot2, a2s, a2d, m2u, s2, 1);
  edge_acc<<<cdiv(tot2, B), B, 0, stream>>>(ei, E, tot2, a2s, a2d, m2u, s2, h2, out, 1, 4);
}
